// LEARNTOBRANCH_Deep_51479478009966
// MI455X (gfx1250) — matrix operand order repaired and checked by reading, not compiled, not run on hardware
//
#include <hip/hip_runtime.h>
#include <hip/hip_bf16.h>
#include <math.h>

typedef __bf16 bf16_t;
typedef __attribute__((ext_vector_type(16))) __bf16 v16bf;
typedef __attribute__((ext_vector_type(8)))  __bf16 v8bf;
typedef __attribute__((ext_vector_type(8)))  float  v8f;

#define BATCH 16
#define NATTR 40

// ---------------------------------------------------------------------------
// utility kernels
// ---------------------------------------------------------------------------
__global__ void k_zero_f32(float* __restrict__ p, long n) {
    long i = (long)blockIdx.x * blockDim.x + threadIdx.x;
    if (i < n) p[i] = 0.0f;
}

// ---------------------------------------------------------------------------
// Weight pre-swizzle: fp32 OIHW [P][COUT][64][3][3] -> bf16 WMMA-A-fragment
// layout [P][COUT/16][chunk=18][lane=32][elem=16], so each lane's A fragment
// for a K-chunk is 32 contiguous bytes.
//   chunk = tap*2 + cb ; lane = grp*16 + m ; cout = ct*16 + m
//   K(e) = 16*grp + e ; cin = cb*32 + K
// ---------------------------------------------------------------------------
__global__ void k_swizzle_w(const float* __restrict__ w, bf16_t* __restrict__ out,
                            int P, int COUT) {
    int nCT = COUT / 16;
    long total = (long)P * nCT * 18 * 32 * 16;
    long i = (long)blockIdx.x * blockDim.x + threadIdx.x;
    if (i >= total) return;
    int e = (int)(i & 15); long r = i >> 4;
    int lane = (int)(r & 31); r >>= 5;
    int chunk = (int)(r % 18); r /= 18;
    int ct = (int)(r % nCT); int p = (int)(r / nCT);
    int tap = chunk >> 1, cb = chunk & 1;
    int grp = lane >> 4, m = lane & 15;
    int K = 16 * grp + e;
    int c = cb * 32 + K;
    int cout = ct * 16 + m;
    out[i] = (bf16_t)w[((long)(p * COUT + cout) * 64 + c) * 9 + tap];
}

// ---------------------------------------------------------------------------
// stem conv: 7x7, pad 0, Cin=3, Cout=64.  x: [16,3,256,256] f32 NCHW
// out: [16,250,250,64] bf16 NHWC  (bias skipped: cancelled exactly by BN)
// ---------------------------------------------------------------------------
__global__ __launch_bounds__(256) void k_conv1(const float* __restrict__ x,
                                               const float* __restrict__ w,
                                               bf16_t* __restrict__ out) {
    const int HW = 250 * 250;
    int idx = blockIdx.x * blockDim.x + threadIdx.x;   // over BATCH*HW
    if (idx >= BATCH * HW) return;
    int b = idx / HW, q = idx % HW;
    int oy = q / 250, ox = q % 250;
    float acc[64];
    #pragma unroll
    for (int o = 0; o < 64; ++o) acc[o] = 0.0f;
    for (int c = 0; c < 3; ++c) {
        for (int ky = 0; ky < 7; ++ky) {
            const float* xr = x + (((long)b * 3 + c) * 256 + (oy + ky)) * 256 + ox;
            for (int kx = 0; kx < 7; ++kx) {
                float v = xr[kx];
                const float* wp = w + (c * 7 + ky) * 7 + kx;
                #pragma unroll
                for (int o = 0; o < 64; ++o) acc[o] += v * wp[o * 147];
            }
        }
    }
    bf16_t* op = out + (long)idx * 64;
    #pragma unroll
    for (int o = 0; o < 64; ++o) op[o] = (bf16_t)acc[o];
}

// ---------------------------------------------------------------------------
// per-(parent,channel) sum / sumsq over bf16 NHWC buffer
// slot = (i / ppe) * C + (i % C)
// ---------------------------------------------------------------------------
__global__ void k_stats_bf16(const bf16_t* __restrict__ buf, float* __restrict__ stats,
                             long n, long ppe, int C, int slots) {
    extern __shared__ float ls[];          // slots*2 floats
    for (int i = threadIdx.x; i < slots * 2; i += blockDim.x) ls[i] = 0.0f;
    __syncthreads();
    long stride = (long)gridDim.x * blockDim.x;
    for (long i = (long)blockIdx.x * blockDim.x + threadIdx.x; i < n; i += stride) {
        float v = (float)buf[i];
        int slot = (int)(i / ppe) * C + (int)(i % (long)C);
        atomicAdd(&ls[slot * 2 + 0], v);
        atomicAdd(&ls[slot * 2 + 1], v * v);
    }
    __syncthreads();
    for (int i = threadIdx.x; i < slots * 2; i += blockDim.x)
        atomicAdd(&stats[i], ls[i]);
}

// in-place BN (training stats) + ReLU on bf16 NHWC buffer
__global__ void k_bnrelu(bf16_t* __restrict__ buf, const float* __restrict__ stats,
                         long n, long ppe, int C, float invCnt) {
    long i = (long)blockIdx.x * blockDim.x + threadIdx.x;
    if (i >= n) return;
    int slot = (int)(i / ppe) * C + (int)(i % (long)C);
    float mean = stats[slot * 2 + 0] * invCnt;
    float var  = stats[slot * 2 + 1] * invCnt - mean * mean;
    float v = ((float)buf[i] - mean) * rsqrtf(var + 1e-5f);
    buf[i] = (bf16_t)(v > 0.0f ? v : 0.0f);
}

// 2x2 maxpool (stride 2, floor) NHWC bf16 -> NHWC bf16, single tensor
__global__ void k_pool(const bf16_t* __restrict__ in, bf16_t* __restrict__ out,
                       int H, int W, int C) {
    int H2 = H / 2, W2 = W / 2;
    long total = (long)BATCH * H2 * W2 * C;
    long i = (long)blockIdx.x * blockDim.x + threadIdx.x;
    if (i >= total) return;
    int c = (int)(i % C); long r = i / C;
    int x2 = (int)(r % W2); r /= W2;
    int y2 = (int)(r % H2); int b = (int)(r / H2);
    const bf16_t* p0 = in + (((long)b * H + 2 * y2) * W + 2 * x2) * C + c;
    float m = fmaxf(fmaxf((float)p0[0], (float)p0[C]),
                    fmaxf((float)p0[(long)W * C], (float)p0[(long)W * C + C]));
    out[i] = (bf16_t)m;
}

// ---------------------------------------------------------------------------
// 3x3 conv, pad 1, Cin=64, implicit GEMM via v_wmma_f32_16x16x32_bf16.
// in  : [P?][B][H][W][64] bf16 NHWC (parent stride in elements; 0 => shared)
// wswz: pre-swizzled weights (see k_swizzle_w)
// out : [P][B][H][W][COUT] bf16 NHWC (raw pre-BN)
// grid: x = ceil(H*W/256), y = batch, z = p*(COUT/16) + coutTile ; 256 thr (8 waves)
// Each wave computes 16 couts x 32 positions (two N tiles sharing each A frag).
// ---------------------------------------------------------------------------
template <int COUT>
__global__ __launch_bounds__(256) void k_conv3x3(const bf16_t* __restrict__ in,
                                                 const bf16_t* __restrict__ wswz,
                                                 bf16_t* __restrict__ out,
                                                 int H, int W, long pstride_in) {
    __shared__ __align__(32) bf16_t ldsw[18 * 32 * 16];   // 18 KB, fragment-major
    const int HW = H * W;
    const int b  = blockIdx.y;
    const int nCT = COUT / 16;
    const int p  = blockIdx.z / nCT;
    const int ct = blockIdx.z % nCT;

    // stage this (p, coutTile)'s swizzled weights: flat contiguous copy
    {
        const v8bf* src = (const v8bf*)(wswz + ((long)(p * nCT + ct)) * 18 * 32 * 16);
        v8bf* dst = (v8bf*)ldsw;
        for (int idx = threadIdx.x; idx < 18 * 32 * 16 / 8; idx += blockDim.x)
            dst[idx] = src[idx];
    }
    __syncthreads();

    const int wave = threadIdx.x >> 5;
    const int lane = threadIdx.x & 31;
    const int grp  = lane >> 4;            // 0 or 1
    const int ln16 = lane & 15;
    const int q0 = blockIdx.x * 256 + wave * 32 + ln16;   // N tile 0 position
    const int q1 = q0 + 16;                               // N tile 1 position
    const int oy0 = q0 / W, ox0 = q0 % W;
    const int oy1 = q1 / W, ox1 = q1 % W;
    // B fragment: lane holds cin = cb*32 + 16*grp + e at column N = ln16
    const bf16_t* ip = in + (long)p * pstride_in + ((long)b * HW) * 64 + 16 * grp;

    v8f acc0 = {0.f, 0.f, 0.f, 0.f, 0.f, 0.f, 0.f, 0.f};
    v8f acc1 = {0.f, 0.f, 0.f, 0.f, 0.f, 0.f, 0.f, 0.f};
    v16bf zero;
    #pragma unroll
    for (int e = 0; e < 16; ++e) zero[e] = (bf16_t)0.0f;

    for (int tap = 0; tap < 9; ++tap) {
        const int dy = tap / 3 - 1, dx = tap % 3 - 1;
        const int yy0 = oy0 + dy, xx0 = ox0 + dx;
        const int yy1 = oy1 + dy, xx1 = ox1 + dx;
        const bool ok0 = (q0 < HW) && (yy0 >= 0) && (yy0 < H) && (xx0 >= 0) && (xx0 < W);
        const bool ok1 = (q1 < HW) && (yy1 >= 0) && (yy1 < H) && (xx1 >= 0) && (xx1 < W);
        const bf16_t* s0 = ip + ((long)yy0 * W + xx0) * 64;
        const bf16_t* s1 = ip + ((long)yy1 * W + xx1) * 64;
        #pragma unroll
        for (int cb = 0; cb < 2; ++cb) {
            const int chunk = tap * 2 + cb;
            v16bf afrag = *(const v16bf*)(ldsw + ((chunk * 32 + lane) << 4));
            v16bf b0, b1;
            if (ok0) b0 = *(const v16bf*)(s0 + cb * 32); else b0 = zero;
            if (ok1) b1 = *(const v16bf*)(s1 + cb * 32); else b1 = zero;
            acc0 = __builtin_amdgcn_wmma_f32_16x16x32_bf16(
                       false, afrag, false, b0, (short)0, acc0, false, false);
            acc1 = __builtin_amdgcn_wmma_f32_16x16x32_bf16(
                       false, afrag, false, b1, (short)0, acc1, false, false);
        }
    }

    // D element v: cout = ct*16 + 8*grp + v, column = q
    if (q0 < HW) {
        bf16_t* op = out + ((((long)p * BATCH + b) * HW + q0) * COUT) + ct * 16 + 8 * grp;
        v8bf ov;
        #pragma unroll
        for (int v = 0; v < 8; ++v) ov[v] = (bf16_t)acc0[v];
        *(v8bf*)op = ov;
    }
    if (q1 < HW) {
        bf16_t* op = out + ((((long)p * BATCH + b) * HW + q1) * COUT) + ct * 16 + 8 * grp;
        v8bf ov;
        #pragma unroll
        for (int v = 0; v < 8; ++v) ov[v] = (bf16_t)acc1[v];
        *(v8bf*)op = ov;
    }
}

// ---------------------------------------------------------------------------
// d = softmax(log(softmax(br/0.5)) / t, axis=parents)   br: [Cn][P]
// ---------------------------------------------------------------------------
__global__ void k_compute_d(const float* __restrict__ br, float* __restrict__ d,
                            int Cn, int P, const int* __restrict__ tptr) {
    int n = blockIdx.x * blockDim.x + threadIdx.x;
    if (n >= Cn) return;
    float t = (float)(*tptr);
    float s[16], u[16];
    float mx = -1e30f;
    for (int p = 0; p < P; ++p) { s[p] = br[n * P + p] * 2.0f; mx = fmaxf(mx, s[p]); }
    float sum = 0.0f;
    for (int p = 0; p < P; ++p) sum += expf(s[p] - mx);
    float lse = logf(sum);
    float mx2 = -1e30f;
    for (int p = 0; p < P; ++p) { u[p] = (s[p] - mx - lse) / t; mx2 = fmaxf(mx2, u[p]); }
    float sum2 = 0.0f;
    for (int p = 0; p < P; ++p) { u[p] = expf(u[p] - mx2); sum2 += u[p]; }
    float inv = 1.0f / sum2;
    for (int p = 0; p < P; ++p) d[n * P + p] = u[p] * inv;
}

// maxpool 2x2 then weighted sum over parents
template <typename OUT_T>
__global__ void k_pool_mix(const bf16_t* __restrict__ act, const float* __restrict__ d,
                           OUT_T* __restrict__ out, int P, int Cn, int H, int W, int C) {
    int H2 = H / 2, W2 = W / 2;
    long total = (long)Cn * BATCH * H2 * W2 * C;
    long i = (long)blockIdx.x * blockDim.x + threadIdx.x;
    if (i >= total) return;
    int c = (int)(i % C); long r = i / C;
    int x2 = (int)(r % W2); r /= W2;
    int y2 = (int)(r % H2); r /= H2;
    int b  = (int)(r % BATCH); int n = (int)(r / BATCH);
    float s = 0.0f;
    for (int p = 0; p < P; ++p) {
        const bf16_t* base = act + (((long)p * BATCH + b) * H * W) * C;
        const bf16_t* r0 = base + ((long)(2 * y2) * W + 2 * x2) * C + c;
        float m = fmaxf(fmaxf((float)r0[0], (float)r0[C]),
                        fmaxf((float)r0[(long)W * C], (float)r0[(long)W * C + C]));
        s += d[n * P + p] * m;
    }
    out[i] = (OUT_T)s;
}

// reorder features [A][B][49][32] (NHWC) -> [A][B][1568] with f = c*49 + s
__global__ void k_reorder_f(const float* __restrict__ fin, float* __restrict__ fout) {
    long total = (long)NATTR * BATCH * 1568;
    long i = (long)blockIdx.x * blockDim.x + threadIdx.x;
    if (i >= total) return;
    int f = (int)(i % 1568); long r = i / 1568;       // r = a*B + b
    int c = f / 49, s = f % 49;
    fout[i] = fin[(r * 49 + s) * 32 + c];
}

// batched MLP head: out[a][b][j] = act( sum_f in[a][b][f]*w[a][f][j] + bias[a][j] )
__global__ void k_fc(const float* __restrict__ in, const float* __restrict__ w,
                     const float* __restrict__ bias, float* __restrict__ out,
                     int Fin, int Fout, int do_relu) {
    long total = (long)NATTR * BATCH * Fout;
    long i = (long)blockIdx.x * blockDim.x + threadIdx.x;
    if (i >= total) return;
    int j = (int)(i % Fout); long r = i / Fout;
    int b = (int)(r % BATCH); int a = (int)(r / BATCH);
    const float* ip = in + ((long)a * BATCH + b) * Fin;
    const float* wp = w + (long)a * Fin * Fout + j;
    float s = bias[a * Fout + j];
    for (int f = 0; f < Fin; ++f) s += ip[f] * wp[(long)f * Fout];
    if (do_relu) s = fmaxf(s, 0.0f);
    out[i] = s;
}

// final head (Fout=1) writing transposed output [B][A]
__global__ void k_fc3(const float* __restrict__ h2, const float* __restrict__ w,
                      const float* __restrict__ bias, float* __restrict__ out) {
    int i = blockIdx.x * blockDim.x + threadIdx.x;   // A*B
    if (i >= NATTR * BATCH) return;
    int b = i % BATCH, a = i / BATCH;
    const float* ip = h2 + ((long)a * BATCH + b) * 128;
    const float* wp = w + (long)a * 128;
    float s = bias[a];
    for (int j = 0; j < 128; ++j) s += ip[j] * wp[j];
    out[b * NATTR + a] = s;
}

// ---------------------------------------------------------------------------
// host orchestration
// ---------------------------------------------------------------------------
static inline long cdiv(long a, long b) { return (a + b - 1) / b; }

extern "C" void kernel_launch(void* const* d_in, const int* in_sizes, int n_in,
                              void* d_out, int out_size, void* d_ws, size_t ws_size,
                              hipStream_t stream) {
    (void)in_sizes; (void)n_in; (void)out_size; (void)ws_size;
    const float* x    = (const float*)d_in[0];
    const float* w10  = (const float*)d_in[1];
    const float* wa[4] = {(const float*)d_in[3],  (const float*)d_in[7],
                          (const float*)d_in[11], (const float*)d_in[15]};
    const float* wb[4] = {(const float*)d_in[5],  (const float*)d_in[9],
                          (const float*)d_in[13], (const float*)d_in[17]};
    const float* br[4] = {(const float*)d_in[19], (const float*)d_in[20],
                          (const float*)d_in[21], (const float*)d_in[22]};
    const float* fw1 = (const float*)d_in[23];
    const float* fb1 = (const float*)d_in[24];
    const float* fw2 = (const float*)d_in[25];
    const float* fb2 = (const float*)d_in[26];
    const float* fw3 = (const float*)d_in[27];
    const float* fb3 = (const float*)d_in[28];
    const int*   tpt = (const int*)d_in[29];
    float* outp = (float*)d_out;

    const int PL[4]  = {2, 4, 8, 16};
    const int CNL[4] = {4, 8, 16, 40};
    const int CBL[4] = {64, 64, 64, 32};
    const int HL[4]  = {125, 62, 31, 15};

    // ---- workspace layout ----
    char* ws = (char*)d_ws;
    size_t off = 0;
    auto take = [&](size_t bytes) -> char* {
        char* r = ws + off;
        off = (off + bytes + 255) & ~(size_t)255;
        return r;
    };
    bf16_t* wabf[4]; bf16_t* wbbf[4];
    for (int L = 0; L < 4; ++L) {
        wabf[L] = (bf16_t*)take((size_t)PL[L] * 64 * 64 * 9 * sizeof(bf16_t));
        wbbf[L] = (bf16_t*)take((size_t)PL[L] * CBL[L] * 64 * 9 * sizeof(bf16_t));
    }
    bf16_t* Y0  = (bf16_t*)take((size_t)BATCH * 250 * 250 * 64 * sizeof(bf16_t)); // 128 MB
    bf16_t* X0  = (bf16_t*)take((size_t)16000000 * sizeof(bf16_t));               // 32 MB
    bf16_t* X1  = (bf16_t*)take((size_t)16000000 * sizeof(bf16_t));               // 32 MB
    bf16_t* BUFA = (bf16_t*)take((size_t)32000000 * sizeof(bf16_t));              // 64 MB
    bf16_t* BUFB = (bf16_t*)take((size_t)32000000 * sizeof(bf16_t));              // 64 MB
    float* STATS = (float*)take(2048 * sizeof(float));
    float* DBUF  = (float*)take(1024 * sizeof(float));
    float* FBUF  = (float*)take((size_t)NATTR * BATCH * 49 * 32 * sizeof(float));
    float* F1    = (float*)take((size_t)NATTR * BATCH * 1568 * sizeof(float));
    float* H1    = (float*)take((size_t)NATTR * BATCH * 128 * sizeof(float));
    float* H2    = (float*)take((size_t)NATTR * BATCH * 128 * sizeof(float));

    // ---- pre-swizzle conv weights into WMMA A-fragment layout (bf16) ----
    for (int L = 0; L < 4; ++L) {
        long na = (long)PL[L] * 64 * 64 * 9;
        long nb = (long)PL[L] * CBL[L] * 64 * 9;
        k_swizzle_w<<<cdiv(na, 256), 256, 0, stream>>>(wa[L], wabf[L], PL[L], 64);
        k_swizzle_w<<<cdiv(nb, 256), 256, 0, stream>>>(wb[L], wbbf[L], PL[L], CBL[L]);
    }

    // ---- stem: conv1 -> BN/ReLU -> pool ----
    {
        long npos = (long)BATCH * 250 * 250;
        k_conv1<<<cdiv(npos, 256), 256, 0, stream>>>(x, w10, Y0);
        long n = npos * 64;
        k_zero_f32<<<1, 256, 0, stream>>>(STATS, 128);
        k_stats_bf16<<<2048, 256, 64 * 2 * sizeof(float), stream>>>(Y0, STATS, n, n, 64, 64);
        k_bnrelu<<<cdiv(n, 256), 256, 0, stream>>>(Y0, STATS, n, n, 64, 1.0f / (float)npos);
        long np = (long)BATCH * 125 * 125 * 64;
        k_pool<<<cdiv(np, 256), 256, 0, stream>>>(Y0, X0, 250, 250, 64);
    }

    // ---- branch layers ----
    bf16_t* curIn = X0;
    bf16_t* curOut = X1;
    for (int L = 0; L < 4; ++L) {
        const int P = PL[L], Cn = CNL[L], Cb = CBL[L], H = HL[L], W = HL[L];
        const long HW = (long)H * W;
        const long pstrideA = (L == 0) ? 0 : (long)BATCH * HW * 64;
        dim3 blk(256);

        // conv_a (64 -> 64), raw into BUFA
        {
            dim3 grd((unsigned)cdiv(HW, 256), BATCH, P * 4);
            k_conv3x3<64><<<grd, blk, 0, stream>>>(curIn, wabf[L], BUFA, H, W, pstrideA);
            long n = (long)P * BATCH * HW * 64, ppe = (long)BATCH * HW * 64;
            int slots = P * 64;
            k_zero_f32<<<cdiv((long)slots * 2, 256) + 1, 256, 0, stream>>>(STATS, slots * 2);
            k_stats_bf16<<<2048, 256, slots * 2 * sizeof(float), stream>>>(BUFA, STATS, n, ppe, 64, slots);
            k_bnrelu<<<cdiv(n, 256), 256, 0, stream>>>(BUFA, STATS, n, ppe, 64, 1.0f / (float)(BATCH * HW));
        }
        // conv_b (64 -> Cb), raw into BUFB
        {
            dim3 grd((unsigned)cdiv(HW, 256), BATCH, P * (Cb / 16));
            long pstrideB = (long)BATCH * HW * 64;
            if (Cb == 64)
                k_conv3x3<64><<<grd, blk, 0, stream>>>(BUFA, wbbf[L], BUFB, H, W, pstrideB);
            else
                k_conv3x3<32><<<grd, blk, 0, stream>>>(BUFA, wbbf[L], BUFB, H, W, pstrideB);
            long n = (long)P * BATCH * HW * Cb, ppe = (long)BATCH * HW * Cb;
            int slots = P * Cb;
            k_zero_f32<<<cdiv((long)slots * 2, 256) + 1, 256, 0, stream>>>(STATS, slots * 2);
            k_stats_bf16<<<2048, 256, slots * 2 * sizeof(float), stream>>>(BUFB, STATS, n, ppe, Cb, slots);
            k_bnrelu<<<cdiv(n, 256), 256, 0, stream>>>(BUFB, STATS, n, ppe, Cb, 1.0f / (float)(BATCH * HW));
        }
        // branching weights + pool/mix
        k_compute_d<<<1, 64, 0, stream>>>(br[L], DBUF, Cn, P, tpt);
        {
            int H2d = H / 2, W2d = W / 2;
            long tot = (long)Cn * BATCH * H2d * W2d * Cb;
            if (L < 3)
                k_pool_mix<bf16_t><<<cdiv(tot, 256), 256, 0, stream>>>(BUFB, DBUF, curOut, P, Cn, H, W, Cb);
            else
                k_pool_mix<float><<<cdiv(tot, 256), 256, 0, stream>>>(BUFB, DBUF, FBUF, P, Cn, H, W, Cb);
        }
        bf16_t* t2 = curIn; curIn = curOut; curOut = t2;
    }

    // ---- attribute heads ----
    {
        long nf = (long)NATTR * BATCH * 1568;
        k_reorder_f<<<cdiv(nf, 256), 256, 0, stream>>>(FBUF, F1);
        long n1 = (long)NATTR * BATCH * 128;
        k_fc<<<cdiv(n1, 256), 256, 0, stream>>>(F1, fw1, fb1, H1, 1568, 128, 1);
        k_fc<<<cdiv(n1, 256), 256, 0, stream>>>(H1, fw2, fb2, H2, 128, 128, 1);
        k_fc3<<<cdiv((long)NATTR * BATCH, 256) + 1, 256, 0, stream>>>(H2, fw3, fb3, outp);
    }
}
